// vq_PoolFormerBlock_20959440404858
// MI455X (gfx1250) — matrix operand order repaired and checked by reading, not compiled, not run on hardware
//
#include <hip/hip_runtime.h>
#include <hip/hip_bf16.h>
#include <math.h>

typedef __attribute__((ext_vector_type(16))) _Float16 v16h;
typedef __attribute__((ext_vector_type(8)))  _Float16 v8h;
typedef __attribute__((ext_vector_type(8)))  float    v8f;
typedef __attribute__((ext_vector_type(16))) float    v16f;

#define BB    32
#define CC    96
#define HH    56
#define WWID  56
#define HWSZ  3136            // 56*56
#define HID   384
#define PSAMP (CC * HWSZ)     // 301056 elements per sample
#define NTOT  ((size_t)BB * CC * HWSZ)   // 9,633,792
#define EPSV  1e-5f

// ---------------------------------------------------------------------------
// Kernel 1/3: per-sample mean and rsqrt(var+eps) (GroupNorm with 1 group).
// 32 blocks (one per sample), 256 threads, f64 accumulation, LDS tree reduce.
// ---------------------------------------------------------------------------
__global__ __launch_bounds__(256)
void stats_kernel(const float* __restrict__ src, float* __restrict__ stats) {
  __shared__ double sSum[256];
  __shared__ double sSq[256];
  const int b = blockIdx.x;
  const int tid = threadIdx.x;
  const float* p = src + (size_t)b * PSAMP;
  double s = 0.0, q = 0.0;
  for (int i = tid; i < PSAMP; i += 256) {
    float v = p[i];
    s += (double)v;
    q += (double)v * (double)v;
  }
  sSum[tid] = s; sSq[tid] = q;
  __syncthreads();
  for (int off = 128; off > 0; off >>= 1) {
    if (tid < off) { sSum[tid] += sSum[tid + off]; sSq[tid] += sSq[tid + off]; }
    __syncthreads();
  }
  if (tid == 0) {
    double mu  = sSum[0] / (double)PSAMP;
    double var = sSq[0] / (double)PSAMP - mu * mu;
    stats[2 * b]     = (float)mu;
    stats[2 * b + 1] = rsqrtf((float)var + EPSV);
  }
}

// ---------------------------------------------------------------------------
// Kernel 2: PoolFormer token mixer.
//   feat1 = x + ls1 * (avgpool3(norm1(x)) - norm1(x))
// norm1 is affine per element (scale/shift constant across the 3x3 window for
// a fixed channel), so pool(norm(v)) = pool(v)*scale + shift — only raw x is
// read for the window. count_include_pad=False handled by the valid count.
// ---------------------------------------------------------------------------
__global__ __launch_bounds__(256)
void mixer_kernel(const float* __restrict__ x, const float* __restrict__ stats1,
                  const float* __restrict__ g1, const float* __restrict__ b1,
                  const float* __restrict__ ls1,
                  float* __restrict__ feat1, float* __restrict__ loss) {
  if (blockIdx.x == 0 && threadIdx.x == 0) loss[0] = 0.0f;
  const size_t total = NTOT;
  const size_t stride = (size_t)gridDim.x * 256u;
  for (size_t idx = (size_t)blockIdx.x * 256u + threadIdx.x; idx < total; idx += stride) {
    int b   = (int)(idx / PSAMP);
    int rem = (int)(idx % PSAMP);
    int c   = rem / HWSZ;
    int hw  = rem % HWSZ;
    int h   = hw / WWID, w = hw % WWID;
    float mu = stats1[2 * b], rs = stats1[2 * b + 1];
    float scale = rs * g1[c];
    float shift = b1[c] - mu * scale;
    const float* plane = x + (size_t)(b * CC + c) * HWSZ;
    float center = plane[hw];
    float sum = 0.0f; int cnt = 0;
    #pragma unroll
    for (int dh = -1; dh <= 1; ++dh) {
      int hhh = h + dh;
      if (hhh < 0 || hhh >= HH) continue;
      #pragma unroll
      for (int dw = -1; dw <= 1; ++dw) {
        int www = w + dw;
        if (www < 0 || www >= WWID) continue;
        sum += plane[hhh * WWID + www];
        ++cnt;
      }
    }
    float pool = (sum / (float)cnt) * scale + shift;
    float xn   = center * scale + shift;
    feat1[idx] = center + ls1[c] * (pool - xn);
  }
}

// ---------------------------------------------------------------------------
// Kernel 4: fused norm2 -> FSQ VQ -> MLP -> residual, WMMA f16/f32.
// Block = 64 threads = 2 waves; block owns 32 pixels of one sample
// (3136 % 32 == 0, so tiles never straddle samples). Each wave computes a
// 16-pixel M-tile with v_wmma_f32_16x16x32_f16.
//
// A-fragment layout (16-bit A 16x32, ISA 7.12.2):
//   element j (VGPR j/2, half j%2) of lane L holds
//   M = L&15,  K = j + 8*(j>=8) + 8*(L>=16)
//   -> two aligned 16B LDS reads per fragment (K runs 0..7 and 16..23 + offs).
// B-fragment: lane L, element j -> N = L&15, K = j + 8*(j>=8) + 8*(L>=16)
//   -> f32 weights K runs 0..7 and 16..23 + offs (4x float4), converted to f16.
// C/D f32 16x16: VGPR r, lane L -> M = r + 8*(L>=16), N = L&15.
// ---------------------------------------------------------------------------
__device__ inline v16h load_a_frag(const _Float16* __restrict__ base) {
  v8h lo = *(const v8h*)base;          // K = +0..7   -> j = 0..7
  v8h hi = *(const v8h*)(base + 16);   // K = +16..23 -> j = 8..15
  return __builtin_shufflevector(lo, hi, 0, 1, 2, 3, 4, 5, 6, 7,
                                         8, 9, 10, 11, 12, 13, 14, 15);
}

__device__ inline v16h load_b_frag(const float* __restrict__ row) {
  const float4* w4 = (const float4*)row;   // 16B-aligned by construction
  float4 f0 = w4[0], f1 = w4[1], f2 = w4[4], f3 = w4[5];
  v16f wf;
  wf[0]  = f0.x; wf[1]  = f0.y; wf[2]  = f0.z; wf[3]  = f0.w;
  wf[4]  = f1.x; wf[5]  = f1.y; wf[6]  = f1.z; wf[7]  = f1.w;
  wf[8]  = f2.x; wf[9]  = f2.y; wf[10] = f2.z; wf[11] = f2.w;
  wf[12] = f3.x; wf[13] = f3.y; wf[14] = f3.z; wf[15] = f3.w;
  return __builtin_convertvector(wf, v16h);
}

__global__ __launch_bounds__(64)
void fused_vq_mlp_kernel(
    const float* __restrict__ feat1, const float* __restrict__ stats2,
    const float* __restrict__ g2,  const float* __restrict__ b2,
    const float* __restrict__ Wp,  const float* __restrict__ bp,
    const float* __restrict__ Wo,  const float* __restrict__ bo,
    const float* __restrict__ Wf1, const float* __restrict__ bf1,
    const float* __restrict__ Wf2, const float* __restrict__ bf2,
    const float* __restrict__ ls2,
    float* __restrict__ out, float* __restrict__ out2) {
  __shared__ float    sXY[32 * 97];    // xn2 as [c*32+p] (3072), later y as [p*97+c]
  __shared__ _Float16 sXQ[32 * 104];   // xq  A-operand staging, stride 104 (bank-safe)
  __shared__ float    sZQ[128];        // quantized codes, 32 pixels x 4 dims
  __shared__ _Float16 sH [32 * 392];   // gelu(h) A-operand staging, stride 392

  const int tid  = threadIdx.x;
  const int lane = tid & 31;
  const int wave = tid >> 5;
  const int pix0 = blockIdx.x * 32;          // global pixel base
  const int b    = pix0 / HWSZ;
  const int hw0  = pix0 % HWSZ;
  const size_t sbase = (size_t)b * CC * HWSZ + hw0;

  const float mu = stats2[2 * b], rs = stats2[2 * b + 1];

  // ---- Stage A: norm2 of feat1 -> sXY[c][p] (coalesced: tid -> p) ----
  for (int i = tid; i < CC * 32; i += 64) {
    int c = i >> 5, p = i & 31;
    float v = feat1[sbase + (size_t)c * HWSZ + p];
    sXY[c * 32 + p] = (v - mu) * rs * g2[c] + b2[c];
  }
  __syncthreads();

  // ---- Stage B: project-in (C->4) + FSQ. levels=3 => zq = round(tanh(z)) ----
  for (int rep = 0; rep < 2; ++rep) {
    int idx = tid + rep * 64;                // 128 = 32 pixels * 4 dims
    int p = idx >> 2, d = idx & 3;
    float acc = bp[d];
    for (int c = 0; c < CC; ++c) acc += sXY[c * 32 + p] * Wp[d * CC + c];
    sZQ[idx] = roundf(tanhf(acc));           // half_l = 1, half_width = 1
  }
  __syncthreads();

  // ---- Stage C: project-out (4->C) -> f16 A staging sXQ[p*104 + c] ----
  for (int i = tid; i < CC * 32; i += 64) {
    int c = i >> 5, p = i & 31;
    float acc = bo[c];
    #pragma unroll
    for (int d = 0; d < 4; ++d) acc += sZQ[p * 4 + d] * Wo[c * 4 + d];
    sXQ[p * 104 + c] = (_Float16)acc;
  }
  __syncthreads();

  const int m      = lane & 15;
  const int hiHalf = lane >> 4;        // lanes 16..31
  const int kofsA  = hiHalf << 3;      // +8 in K for A; +8 in M for C/D
  const int kofsB  = hiHalf << 3;      // +8 in K for B
  const int prow   = wave * 16 + m;    // pixel row this lane contributes to A

  // ---- Stage D: GEMM1 (16x384x96 per wave) + exact GELU -> sH ----
  for (int nt = 0; nt < HID / 16; ++nt) {
    const int o0 = nt * 16;
    const float bias = bf1[o0 + m];    // N = lane&15
    v8f acc;
    #pragma unroll
    for (int r = 0; r < 8; ++r) acc[r] = bias;
    #pragma unroll
    for (int kt = 0; kt < CC / 32; ++kt) {
      v16h a  = load_a_frag(&sXQ[prow * 104 + kt * 32 + kofsA]);
      v16h bm = load_b_frag(Wf1 + (size_t)(o0 + m) * CC + kt * 32 + kofsB);
      acc = __builtin_amdgcn_wmma_f32_16x16x32_f16(false, a, false, bm,
                                                   (short)0, acc, false, false);
    }
    #pragma unroll
    for (int r = 0; r < 8; ++r) {
      float xv = acc[r];
      float g  = 0.5f * xv * (1.0f + erff(xv * 0.70710678118654752f));
      sH[(wave * 16 + r + kofsA) * 392 + o0 + m] = (_Float16)g;
    }
  }
  __syncthreads();

  // ---- Stage E: GEMM2 (16x96x384 per wave) -> y staged in sXY[p*97 + c] ----
  for (int nt = 0; nt < CC / 16; ++nt) {
    const int c0 = nt * 16;
    const float bias = bf2[c0 + m];
    v8f acc;
    #pragma unroll
    for (int r = 0; r < 8; ++r) acc[r] = bias;
    #pragma unroll
    for (int kt = 0; kt < HID / 32; ++kt) {
      v16h a  = load_a_frag(&sH[prow * 392 + kt * 32 + kofsA]);
      v16h bm = load_b_frag(Wf2 + (size_t)(c0 + m) * HID + kt * 32 + kofsB);
      acc = __builtin_amdgcn_wmma_f32_16x16x32_f16(false, a, false, bm,
                                                   (short)0, acc, false, false);
    }
    #pragma unroll
    for (int r = 0; r < 8; ++r)
      sXY[(wave * 16 + r + kofsA) * 97 + c0 + m] = acc[r];
  }
  __syncthreads();

  // ---- Stage F: residual + fully coalesced writeback (out and its dup) ----
  for (int i = tid; i < CC * 32; i += 64) {
    int c = i >> 5, p = i & 31;
    size_t idx = sbase + (size_t)c * HWSZ + p;
    float v = feat1[idx] + ls2[c] * sXY[p * 97 + c];
    out[idx]  = v;
    out2[idx] = v;
  }
}

// ---------------------------------------------------------------------------
// Launch: stats(x) -> mixer -> stats(feat1) -> fused WMMA block.
// d_out layout: [ out (N) | loss (1) | feat1 (N) | out (N) ]   N = 9,633,792
// d_ws: floats [0..63] = (mu,rsig) per sample for norm1; [64..127] for norm2.
// ---------------------------------------------------------------------------
extern "C" void kernel_launch(void* const* d_in, const int* in_sizes, int n_in,
                              void* d_out, int out_size, void* d_ws, size_t ws_size,
                              hipStream_t stream) {
  (void)in_sizes; (void)n_in; (void)out_size; (void)ws_size;
  const float* x   = (const float*)d_in[0];
  const float* g1  = (const float*)d_in[1];
  const float* b1  = (const float*)d_in[2];
  const float* g2  = (const float*)d_in[3];
  const float* b2  = (const float*)d_in[4];
  const float* ls1 = (const float*)d_in[5];
  const float* ls2 = (const float*)d_in[6];
  const float* Wp  = (const float*)d_in[7];
  const float* bp  = (const float*)d_in[8];
  const float* Wo  = (const float*)d_in[9];
  const float* bo  = (const float*)d_in[10];
  const float* Wf1 = (const float*)d_in[11];
  const float* bf1 = (const float*)d_in[12];
  const float* Wf2 = (const float*)d_in[13];
  const float* bf2 = (const float*)d_in[14];

  float* outAll = (float*)d_out;
  float* loss   = outAll + NTOT;
  float* feat1  = outAll + NTOT + 1;
  float* out2   = outAll + 2 * NTOT + 1;
  float* stats  = (float*)d_ws;

  stats_kernel<<<BB, 256, 0, stream>>>(x, stats);
  mixer_kernel<<<2048, 256, 0, stream>>>(x, stats, g1, b1, ls1, feat1, loss);
  stats_kernel<<<BB, 256, 0, stream>>>(feat1, stats + 64);

  const int nPixelTiles = (BB * HWSZ) / 32;   // 3136 blocks of 32 pixels
  fused_vq_mlp_kernel<<<nPixelTiles, 64, 0, stream>>>(
      feat1, stats + 64, g2, b2, Wp, bp, Wo, bo,
      Wf1, bf1, Wf2, bf2, ls2, outAll, out2);
}
